// LWMSA_89601607729351
// MI455X (gfx1250) — matrix operand order repaired and checked by reading, not compiled, not run on hardware
//
#include <hip/hip_runtime.h>
#include <hip/hip_bf16.h>

typedef __attribute__((ext_vector_type(16))) _Float16 v16h;
typedef __attribute__((ext_vector_type(8)))  _Float16 v8h;
typedef __attribute__((ext_vector_type(8)))  float    v8f;
typedef __attribute__((ext_vector_type(2)))  float    v2f;
typedef __attribute__((ext_vector_type(4)))  unsigned int u32x4;
typedef __attribute__((ext_vector_type(8)))  unsigned int u32x8;

#define TPB 256

// Padded LDS strides (coprime-ish with 64 banks to avoid conflicts)
constexpr int QKV_LD = 260;   // f32 elements per row of qkv (192 rows)
constexpr int XT_LD  = 72;    // f16 elements per row of x-window transpose (256 rows)
constexpr int WH_LD  = 72;    // f16 elements per row of W (192 rows)
constexpr int KF_LD  = 260;   // f32 elements per row of KF/VV staging (8 rows)

constexpr size_t OFF_QKV = 0;
constexpr size_t SZ_QKV  = (size_t)192 * QKV_LD * sizeof(float);        // 199680
constexpr size_t OFF_XT  = OFF_QKV + SZ_QKV;
constexpr size_t SZ_XT   = (size_t)256 * XT_LD * sizeof(_Float16);      // 36864
constexpr size_t OFF_WH  = OFF_XT + SZ_XT;
constexpr size_t SZ_WH   = (size_t)192 * WH_LD * sizeof(_Float16);      // 27648
constexpr size_t OFF_KF  = OFF_WH + SZ_WH;
constexpr size_t SZ_KF   = (size_t)8 * KF_LD * sizeof(float);           // 8320
constexpr size_t OFF_VV  = OFF_KF + SZ_KF;
constexpr size_t OFF_KVA = OFF_VV + SZ_KF;
constexpr size_t SZ_KVA  = 256 * sizeof(float);                          // 16x16 f32
constexpr size_t OFF_RED = OFF_KVA + SZ_KVA;
constexpr size_t SMEM_BYTES = OFF_RED + 16 * sizeof(float);             // 281920 B < 320KB

// f32 staging of the x-window loaded by the TDM; aliases the QKV region,
// which is dead until after the f16-conversion barrier.
constexpr size_t OFF_XF32 = OFF_QKV;                                     // 64KB

__device__ __forceinline__ float elu1f(float t) {
  // elu(t)+1 = t+1 (t>0) else exp(t)
  return t > 0.0f ? t + 1.0f : __expf(t);
}

__global__ __launch_bounds__(TPB)
void lwmsa_fused_kernel(const float* __restrict__ x,
                        const float* __restrict__ w,
                        float* __restrict__ out) {
  extern __shared__ char smem[];
  float*    QKV = (float*)(smem + OFF_QKV);       // [192][QKV_LD] f32
  float*    XF32= (float*)(smem + OFF_XF32);      // [64][16][16]  f32 (TDM dest)
  _Float16* Xt  = (_Float16*)(smem + OFF_XT);     // [256][XT_LD]  f16 (pixel-major)
  _Float16* Wh  = (_Float16*)(smem + OFF_WH);     // [192][WH_LD]  f16 (o-major)
  float*    KF  = (float*)(smem + OFF_KF);        // [8][KF_LD]    f32
  float*    VV  = (float*)(smem + OFF_VV);        // [8][KF_LD]    f32
  float*    KVA = (float*)(smem + OFF_KVA);       // [16][16]      f32 kv accumulator
  float*    RED = (float*)(smem + OFF_RED);       // [8]           f32 wave partials

  const int tid  = threadIdx.x;
  const int wave = tid >> 5;
  const int lane = tid & 31;
  const int lm   = lane & 15;   // lane % 16
  const int lh   = lane >> 4;   // lane half

  // window id: blk = (b*32 + wh)*32 + wwi
  const int blk = blockIdx.x;
  const int b   = blk >> 10;
  const int wh  = (blk >> 5) & 31;
  const int wwi = blk & 31;

  // ---- Phase 0a: Tensor Data Mover: x window (64ch x 16 x 16 f32) -> LDS ----
  // 3D tile: dim0 = 16 contiguous f32, dim1 = 16 rows (stride 512),
  //          dim2 = 64 channels (stride 512*512). One DMA per workgroup.
  if (wave == 0) {
    const unsigned long long gaddr =
        (unsigned long long)(const void*)(x + ((((size_t)b * 64) * 512 + wh * 16) * 512 + wwi * 16));
    const unsigned lds_addr =
        (unsigned)(unsigned long long)(const void*)XF32;  // low 32 bits = LDS byte offset

    u32x4 g0;
    g0[0] = 1u;                                   // count=1 (valid), user mode
    g0[1] = lds_addr;                             // LDS dest (bytes)
    g0[2] = (unsigned)gaddr;                      // global_addr[31:0]
    g0[3] = ((unsigned)(gaddr >> 32) & 0x01FFFFFFu) | (2u << 30);  // addr[56:32] | type=2

    u32x8 g1;
    g1[0] = 2u << 16;                             // data_size=2 (4 bytes); no flags
    g1[1] = 512u << 16;                           // tensor_dim0[15:0]=512 in bits[63:48]
    g1[2] = 512u << 16;                           // dim0 hi=0; tensor_dim1[15:0]=512
    g1[3] = 16u << 16;                            // dim1 hi=0; tile_dim0=16
    g1[4] = (64u << 16) | 16u;                    // tile_dim1=16, tile_dim2=64
    g1[5] = 512u;                                 // tensor_dim0_stride lo = 512
    g1[6] = (262144u & 0xFFFFu) << 16;            // dim0_stride hi=0; dim1_stride[15:0]
    g1[7] = 262144u >> 16;                        // tensor_dim1_stride[47:16] = 4

    u32x4 g2;
    g2[0] = 64u;                                  // tensor_dim2 = 64 channels
    g2[1] = 0u;                                   // tensor_dim3 unused
    g2[2] = 262144u;                              // tensor_dim2_stride lo = 512*512
    g2[3] = 0u;                                   // stride hi=0; tile_dim3=0 (unused)

    u32x4 g3;
    g3[0] = 0u; g3[1] = 0u; g3[2] = 0u; g3[3] = 0u;  // dims 3/4 unused

    asm volatile("tensor_load_to_lds %0, %1, %2, %3"
                 :
                 : "s"(g0), "s"(g1), "s"(g2), "s"(g3)
                 : "memory");
    __builtin_amdgcn_s_wait_tensorcnt(0);
  }

  // ---- Phase 0b: W (192x64 f32) -> Wh f16 LDS (overlaps the TDM DMA) ----
  for (int i = tid; i < 192 * 64 / 4; i += TPB) {
    const float4 wv = ((const float4*)w)[i];
    const int o = i >> 4;             // 16 float4 per row of 64
    const int c = (i & 15) << 2;
    _Float16* dst = Wh + o * WH_LD + c;
    dst[0] = (_Float16)wv.x; dst[1] = (_Float16)wv.y;
    dst[2] = (_Float16)wv.z; dst[3] = (_Float16)wv.w;
  }
  __syncthreads();

  // ---- Phase 0c: XF32 [c][r][16] -> Xt f16, transposed [p][c] ----
  for (int i = tid; i < 64 * 16 * 4; i += TPB) {
    const int c = i >> 6, rem = i & 63, r = rem >> 2, g = rem & 3;
    const float4 xv = *(const float4*)(XF32 + ((c * 16 + r) * 16 + g * 4));
    const int p = r * 16 + g * 4;
    Xt[(p + 0) * XT_LD + c] = (_Float16)xv.x;
    Xt[(p + 1) * XT_LD + c] = (_Float16)xv.y;
    Xt[(p + 2) * XT_LD + c] = (_Float16)xv.z;
    Xt[(p + 3) * XT_LD + c] = (_Float16)xv.w;
  }
  __syncthreads();

  // ---- Phase 1: QKV[192][256] = W(192x64) @ Xwin(64x256), f16 WMMA, f32 acc ----
  for (int t = wave; t < 12 * 16; t += 8) {
    const int mt = t >> 4, nt = t & 15;
    const int o  = mt * 16 + lm;      // A row for this lane
    const int pc = nt * 16 + lm;      // B/C column for this lane
    v8f acc = {0.f, 0.f, 0.f, 0.f, 0.f, 0.f, 0.f, 0.f};
#pragma unroll
    for (int ks = 0; ks < 2; ++ks) {
      const int kb = ks * 32;
      // A fragment: 16-bit A 16x32 layout: half lh holds K = kb+8*lh+{0..7} and +16
      const int cA = kb + (lh << 3);
      v8h alo = *(const v8h*)(Wh + o * WH_LD + cA);
      v8h ahi = *(const v8h*)(Wh + o * WH_LD + cA + 16);
      v16h a = __builtin_shufflevector(alo, ahi, 0, 1, 2, 3, 4, 5, 6, 7,
                                       8, 9, 10, 11, 12, 13, 14, 15);
      // B fragment: half lh holds K = kb+8*lh+{0..7} and +16 (same as A)
      const int cB = kb + (lh << 3);
      v8h blo = *(const v8h*)(Xt + pc * XT_LD + cB);
      v8h bhi = *(const v8h*)(Xt + pc * XT_LD + cB + 16);
      v16h bm = __builtin_shufflevector(blo, bhi, 0, 1, 2, 3, 4, 5, 6, 7,
                                        8, 9, 10, 11, 12, 13, 14, 15);
      acc = __builtin_amdgcn_wmma_f32_16x16x32_f16(
          false, a, false, bm, (short)0, acc, false, false);
    }
    // C layout: vgpr rr -> (m = rr + 8*lh, n = lm)
#pragma unroll
    for (int rr = 0; rr < 8; ++rr)
      QKV[(mt * 16 + rr + 8 * lh) * QKV_LD + pc] = acc[rr];
  }
  __syncthreads();

  // ---- Phase 2: sequential linear attention, thread p owns window pixel p ----
  const int r0 = tid >> 4, c0 = tid & 15;
  const int hpix = wh * 16 + r0, wpix = wwi * 16 + c0;
  float feat[8];
#pragma unroll
  for (int i = 0; i < 8; ++i) feat[i] = 0.f;

  for (int head = 0; head < 8; ++head) {
    float qf[8], kf[8], vv[8];
    float sq = 0.f, sk = 0.f;
#pragma unroll
    for (int dch = 0; dch < 8; ++dch) {
      const int row = head * 8 + dch;
      float fq = QKV[row * QKV_LD + tid];
      float fk = QKV[(64 + row) * QKV_LD + tid];
      float fv = QKV[(128 + row) * QKV_LD + tid];
      if (head > 0) { fq += feat[dch]; fk += feat[dch]; fv += feat[dch]; }
      qf[dch] = elu1f(fq);
      kf[dch] = elu1f(fk);
      vv[dch] = fv;
      sq += qf[dch];
      sk += kf[dch];
      KF[dch * KF_LD + tid] = kf[dch];
      VV[dch * KF_LD + tid] = vv[dch];
    }
    KVA[tid] = 0.f;
    // Reduce total K-feature sum S across workgroup (butterfly + LDS partials)
#pragma unroll
    for (int off = 16; off > 0; off >>= 1)
      sk += __shfl_xor(sk, off, 32);
    if (lane == 0) RED[wave] = sk;
    __syncthreads();
    float S = 8.0f * 1e-6f;  // d * EPS folded in
#pragma unroll
    for (int wv2 = 0; wv2 < 8; ++wv2) S += RED[wv2];

    // kv(8x8) = KF(8x256) @ VV(8x256)^T via f32 WMMA 16x16x4; K sliced per wave
    {
      v8f c = {0.f, 0.f, 0.f, 0.f, 0.f, 0.f, 0.f, 0.f};
#pragma unroll
      for (int ks = 0; ks < 8; ++ks) {
        const int pb = wave * 32 + ks * 4 + 2 * lh;  // K (pixel) base for lane
        float a0 = 0.f, a1 = 0.f, b0 = 0.f, b1 = 0.f;
        if (lm < 8) {  // rows/cols 8..15 zero-padded
          a0 = KF[lm * KF_LD + pb];
          a1 = KF[lm * KF_LD + pb + 1];
          b0 = VV[lm * KF_LD + pb];
          b1 = VV[lm * KF_LD + pb + 1];
        }
        v2f av = {a0, a1};
        v2f bv = {b0, b1};
        c = __builtin_amdgcn_wmma_f32_16x16x4_f32(
            false, av, false, bv, (short)0, c, false, false);
      }
      // valid 8x8 region sits in lh==0 lanes; combine wave partials via LDS atomics
      if (lh == 0 && lm < 8) {
#pragma unroll
        for (int rr = 0; rr < 8; ++rr)
          atomicAdd(&KVA[rr * 16 + lm], c[rr]);
      }
    }
    __syncthreads();

    const float z = 1.0f / (sq * S);
#pragma unroll
    for (int cc = 0; cc < 8; ++cc) {
      float acc2 = 0.f;
#pragma unroll
      for (int dch = 0; dch < 8; ++dch)
        acc2 += qf[dch] * KVA[dch * 16 + cc];
      feat[cc] = acc2 * z;
    }
#pragma unroll
    for (int cc = 0; cc < 8; ++cc)
      out[(((size_t)b * 64 + head * 8 + cc) * 512 + hpix) * 512 + wpix] = feat[cc];
    __syncthreads();  // protect KF/VV/KVA before next head rewrites them
  }
}

extern "C" void kernel_launch(void* const* d_in, const int* in_sizes, int n_in,
                              void* d_out, int out_size, void* d_ws, size_t ws_size,
                              hipStream_t stream) {
  const float* x = (const float*)d_in[0];   // (4, 64, 512, 512) f32
  const float* w = (const float*)d_in[1];   // (192, 64) f32
  float* out = (float*)d_out;               // (4, 64, 512, 512) f32

  dim3 grid(4 * 32 * 32);                   // one workgroup per window
  dim3 block(TPB);
  lwmsa_fused_kernel<<<grid, block, SMEM_BYTES, stream>>>(x, w, out);
}
